// Network_80985903334261
// MI455X (gfx1250) — compile-verified, matrix operand order checked by reading, not run on hardware
//
#include <hip/hip_runtime.h>

// ---------------------------------------------------------------------------
// CDNA5 (gfx1250) fused equivariant-GNN forward.
//   - radial MLP GEMMs on v_wmma_f32_16x16x32_f16 (f32 accumulate)
//   - CG tensor product + scatter fused per 64-edge tile (w never hits HBM)
// ---------------------------------------------------------------------------

typedef __attribute__((ext_vector_type(16))) _Float16 v16h;
typedef __attribute__((ext_vector_type(8)))  _Float16 v8h;
typedef __attribute__((ext_vector_type(8)))  float    v8f;

#define DEV __device__ __forceinline__

// problem constants
constexpr int   MUL      = 10;
constexpr int   NGRAPH   = 64;
constexpr int   N_NODES  = 20000;
constexpr int   N_EDGES  = 400000;
constexpr int   RAD_H    = 200;
constexpr int   KP       = 224;     // K padded to multiple of 32
constexpr int   EPW      = 64;      // edges per workgroup (4 wave32 M-tiles)
constexpr float SIGMA    = 5.0f;    // CUTOFF/(NG-1)
constexpr float SQ2      = 0.70710678118654752f;
constexpr float INV_SQRT10 = 0.31622776601683794f;
constexpr float INV_SQRT20 = 0.22360679774997896f;  // == 1/sqrt(NNORM)
constexpr float INV_SQRT3  = 0.57735026918962576f;
constexpr float SQRT3      = 1.7320508075688772f;

DEV float sigmoidf(float x) { return 1.0f / (1.0f + __expf(-x)); }

// ---------------------------------------------------------------------------
// weight packer: fp32 [K][Nsrc] row-major  ->  f16 [Npad][KP]  (B as N x K)
// ---------------------------------------------------------------------------
__global__ void k_pack_w(const float* __restrict__ W, _Float16* __restrict__ Wt,
                         int K, int Nsrc, int Npad) {
    int idx = blockIdx.x * blockDim.x + threadIdx.x;
    if (idx >= Npad * KP) return;
    int n = idx / KP;
    int k = idx - n * KP;
    float v = (n < Nsrc && k < K) ? W[k * Nsrc + n] : 0.0f;
    Wt[idx] = (_Float16)v;
}

__global__ void k_zero(float* __restrict__ p, int n) {
    int i = blockIdx.x * blockDim.x + threadIdx.x;
    if (i < n) p[i] = 0.0f;
}

__global__ void k_node_init(const int* __restrict__ z, const float* __restrict__ emb,
                            float* __restrict__ s_cur) {
    int n = blockIdx.x * blockDim.x + threadIdx.x;
    if (n >= N_NODES) return;
    int zi = z[n];
#pragma unroll
    for (int i = 0; i < MUL; ++i) s_cur[n * MUL + i] = emb[zi * MUL + i];
}

// ---------------------------------------------------------------------------
// fused per-edge-tile kernel:
//   geometry -> rbf -> h0 (VALU, K=3) -> h1 = WMMA(200x200) -> w = WMMA(200xNW)
//   -> CG tensor product messages -> atomic scatter to node aggregates.
// segments of w:  A [ms*out_s] | B [mv*out_s] | C [ms*out_v] | D [mv*out_v]
// ---------------------------------------------------------------------------
template <int MS, int MV, int OUT_S, int OUT_V, int NW, int NCHUNK, int NMS, int NMV>
__global__ void __launch_bounds__(128)
k_edge_fused(const float* __restrict__ pos, const int* __restrict__ eidx,
             const float* __restrict__ s_cur, const float* __restrict__ v_cur,
             const float* __restrict__ W0, const float* __restrict__ b0,
             const _Float16* __restrict__ W1t, const float* __restrict__ b1,
             const _Float16* __restrict__ W2t, const float* __restrict__ b2,
             float* __restrict__ agg_s, float* __restrict__ agg_v) {
    constexpr bool HAS_V = (MV > 0);
    constexpr int  MVp   = HAS_V ? MV : 1;
    constexpr int  OVp   = (OUT_V > 0) ? OUT_V : 1;
    constexpr int  AEND  = MS * OUT_S;
    constexpr int  BEND  = AEND + (HAS_V ? MV * OUT_S : 0);
    constexpr int  CEND  = BEND + MS * OUT_V;
    constexpr int  DEND  = CEND + (HAS_V ? MV * OUT_V : 0);
    static_assert(DEND == NW, "segment layout mismatch");
    constexpr float INV_NMS = (NMS == 2) ? SQ2 : 1.0f;
    constexpr float INV_NMV = (NMV == 2) ? SQ2 : 1.0f;

    extern __shared__ __align__(16) char smem[];
    _Float16* hA  = (_Float16*)smem;                    // [EPW][KP] f16
    _Float16* hB  = hA + EPW * KP;                      // [EPW][KP] f16
    float*    wch = (float*)(hB + EPW * KP);            // [EPW][64] w chunk
    float*    sAl = wch + EPW * 64;                     // [EPW][MS]
    float*    dBl = sAl + EPW * MS;                     // [EPW][MVp]
    float*    sCl = dBl + EPW * MVp;                    // [EPW][MS]
    float*    vDl = sCl + EPW * MS;                     // [EPW][MVp*3]
    float*    msgS = vDl + EPW * MVp * 3;               // [EPW][OUT_S]
    float*    cCl = msgS + EPW * OUT_S;                 // [EPW][OVp]
    float*    mDl = cCl + EPW * OVp;                    // [EPW][OVp*3]
    float*    dsl = mDl + EPW * OVp * 3;                // [EPW] edge length

    const int t  = threadIdx.x;
    const int wv = t >> 5;
    const int L  = t & 31;

    int   colIdx  = 0;
    float sh1r[3] = {0.f, 0.f, 0.f};

    // ---------------- phase 1: geometry, gathers, pre-scaled TP operands ----
    if (t < EPW) {
        const int e = blockIdx.x * EPW + t;
        const int r = eidx[e];
        const int c = eidx[N_EDGES + e];
        colIdx = c;
        float ex = pos[3 * r + 0] - pos[3 * c + 0];
        float ey = pos[3 * r + 1] - pos[3 * c + 1];
        float ez = pos[3 * r + 2] - pos[3 * c + 2];
        float dd = sqrtf(ex * ex + ey * ey + ez * ez + 1e-12f);
        dsl[t] = dd;
        float sc = SQRT3 * INV_SQRT20 / dd;     // sh1 = sqrt3 * unit * 1/sqrt(NNORM)
        sh1r[0] = ex * sc; sh1r[1] = ey * sc; sh1r[2] = ez * sc;
#pragma unroll
        for (int i = 0; i < MS; ++i) {
            float sj = s_cur[r * MS + i];
            sAl[t * MS + i] = sj * (INV_SQRT20 * INV_SQRT10 * INV_NMS);  // *sh0
            sCl[t * MS + i] = sj * (INV_SQRT10 * INV_NMV);
        }
        if (HAS_V) {
#pragma unroll
            for (int i = 0; i < MVp; ++i) {
                float vx = v_cur[(r * MVp + i) * 3 + 0];
                float vy = v_cur[(r * MVp + i) * 3 + 1];
                float vz = v_cur[(r * MVp + i) * 3 + 2];
                dBl[t * MVp + i] = (vx * sh1r[0] + vy * sh1r[1] + vz * sh1r[2]) *
                                   (INV_SQRT3 * INV_SQRT10 * INV_NMS);
                float sd = INV_SQRT20 * INV_SQRT10 * INV_NMV;            // sh0/sqrt(mv)
                vDl[(t * MVp + i) * 3 + 0] = vx * sd;
                vDl[(t * MVp + i) * 3 + 1] = vy * sd;
                vDl[(t * MVp + i) * 3 + 2] = vz * sd;
            }
        }
#pragma unroll
        for (int o = 0; o < OUT_S; ++o) msgS[t * OUT_S + o] = 0.f;
        if (OUT_V > 0) {
#pragma unroll
            for (int o = 0; o < OVp; ++o) cCl[t * OVp + o] = 0.f;
#pragma unroll
            for (int q = 0; q < OVp * 3; ++q) mDl[t * OVp * 3 + q] = 0.f;
        }
    }
    __syncthreads();

    // ---------------- phase 1b: h0 = swish(rbf @ W0 + b0), K=3 on VALU ------
    {
        const int e    = t >> 1;
        const int half = t & 1;
        float dd = dsl[e];
        float u0 = dd * (1.0f / SIGMA);
        float u1 = (dd - 5.0f) * (1.0f / SIGMA);
        float u2 = (dd - 10.0f) * (1.0f / SIGMA);
        float g0 = __expf(-u0 * u0), g1 = __expf(-u1 * u1), g2 = __expf(-u2 * u2);
        const int nb = half * 100;
        for (int n = nb; n < nb + 100; ++n) {
            float a = b0[n] + g0 * W0[n] + g1 * W0[RAD_H + n] + g2 * W0[2 * RAD_H + n];
            hA[e * KP + n] = (_Float16)(a * sigmoidf(a));
        }
        if (half) { for (int p = RAD_H; p < KP; ++p) hA[e * KP + p] = (_Float16)0.f; }
        else      { for (int p = 208;   p < KP; ++p) hB[e * KP + p] = (_Float16)0.f; }
    }
    __syncthreads();

    // fragment geometry (ISA 16-bit A 16x32 / f32 D layouts)
    const int Mrow = wv * 16 + (L & 15);       // A-matrix row (edge)
    const int koff = (L >> 4) << 3;            // +8 K for upper half-wave
    const int rup  = (L >> 4) << 3;            // +8 M for upper half-wave (D)
    const int ncl  = L & 15;                   // D / B column

    // ---------------- GEMM2: h1 = swish(h0 @ W1 + b1) -----------------------
    for (int nt = 0; nt < 13; ++nt) {
        const int ncol = nt * 16 + ncl;
        v8f acc = {0.f, 0.f, 0.f, 0.f, 0.f, 0.f, 0.f, 0.f};
#pragma unroll
        for (int kt = 0; kt < 7; ++kt) {
            const int kb = kt * 32 + koff;
            v8h a0 = *(const v8h*)&hA[Mrow * KP + kb];
            v8h a1 = *(const v8h*)&hA[Mrow * KP + kb + 16];
            v16h A = __builtin_shufflevector(a0, a1, 0, 1, 2, 3, 4, 5, 6, 7,
                                             8, 9, 10, 11, 12, 13, 14, 15);
            const _Float16* bp = W1t + ncol * KP + kb;
            v8h b0v = *(const v8h*)bp;
            v8h b1v = *(const v8h*)(bp + 16);
            v16h B = __builtin_shufflevector(b0v, b1v, 0, 1, 2, 3, 4, 5, 6, 7,
                                             8, 9, 10, 11, 12, 13, 14, 15);
            acc = __builtin_amdgcn_wmma_f32_16x16x32_f16(false, A, false, B,
                                                         (short)0, acc, false, false);
        }
        const float bias = (ncol < RAD_H) ? b1[ncol] : 0.f;
#pragma unroll
        for (int rr = 0; rr < 8; ++rr) {
            const int er = wv * 16 + rr + rup;
            float x = acc[rr] + bias;
            hB[er * KP + ncol] = (_Float16)(x * sigmoidf(x));
        }
    }
    __syncthreads();

    // ---------------- GEMM3 (w chunks) fused with tensor-product consume ----
    for (int ch = 0; ch < NCHUNK; ++ch) {
        if (ch + 1 < NCHUNK)
            __builtin_prefetch(W2t + (ch + 1) * 64 * KP + L * 2 * KP, 0, 0);
#pragma unroll
        for (int ntl = 0; ntl < 4; ++ntl) {
            const int ng = ch * 64 + ntl * 16 + ncl;   // global output index
            v8f acc = {0.f, 0.f, 0.f, 0.f, 0.f, 0.f, 0.f, 0.f};
#pragma unroll
            for (int kt = 0; kt < 7; ++kt) {
                const int kb = kt * 32 + koff;
                v8h a0 = *(const v8h*)&hB[Mrow * KP + kb];
                v8h a1 = *(const v8h*)&hB[Mrow * KP + kb + 16];
                v16h A = __builtin_shufflevector(a0, a1, 0, 1, 2, 3, 4, 5, 6, 7,
                                                 8, 9, 10, 11, 12, 13, 14, 15);
                const _Float16* bp = W2t + ng * KP + kb;
                v8h b0v = *(const v8h*)bp;
                v8h b1v = *(const v8h*)(bp + 16);
                v16h B = __builtin_shufflevector(b0v, b1v, 0, 1, 2, 3, 4, 5, 6, 7,
                                                 8, 9, 10, 11, 12, 13, 14, 15);
                acc = __builtin_amdgcn_wmma_f32_16x16x32_f16(false, A, false, B,
                                                             (short)0, acc, false, false);
            }
            const float bias = (ng < NW) ? b2[ng] : 0.f;
#pragma unroll
            for (int rr = 0; rr < 8; ++rr)
                wch[(wv * 16 + rr + rup) * 64 + ntl * 16 + ncl] = acc[rr] + bias;
        }
        __syncthreads();

        if (t < EPW) {
            for (int j = 0; j < 64; ++j) {          // jj is wave-uniform
                const int jj = ch * 64 + j;
                const float wv_ = wch[t * 64 + j];
                if (jj < AEND) {
                    int i = jj / OUT_S, o = jj - i * OUT_S;
                    msgS[t * OUT_S + o] += sAl[t * MS + i] * wv_;
                } else if (HAS_V && jj < BEND) {
                    int q = jj - AEND, i = q / OUT_S, o = q - i * OUT_S;
                    msgS[t * OUT_S + o] += dBl[t * MVp + i] * wv_;
                } else if (OUT_V > 0 && jj < CEND) {
                    int q = jj - BEND, i = q / OVp, o = q - i * OVp;
                    cCl[t * OVp + o] += sCl[t * MS + i] * wv_;
                } else if (OUT_V > 0 && HAS_V && jj < DEND) {
                    int q = jj - CEND, i = q / OVp, o = q - i * OVp;
                    const float* vd = &vDl[(t * MVp + i) * 3];
                    float* md = &mDl[(t * OVp + o) * 3];
                    md[0] += vd[0] * wv_; md[1] += vd[1] * wv_; md[2] += vd[2] * wv_;
                }
            }
        }
        __syncthreads();
    }

    // ---------------- scatter-add messages to node aggregates ---------------
    if (t < EPW) {
#pragma unroll
        for (int o = 0; o < OUT_S; ++o)
            atomicAdd(&agg_s[colIdx * OUT_S + o], msgS[t * OUT_S + o]);
        if (OUT_V > 0) {
#pragma unroll
            for (int o = 0; o < OUT_V; ++o) {
                const float cc = cCl[t * OVp + o];
#pragma unroll
                for (int m = 0; m < 3; ++m)
                    atomicAdd(&agg_v[(colIdx * OUT_V + o) * 3 + m],
                              cc * sh1r[m] + mDl[(t * OVp + o) * 3 + m]);
            }
        }
    }
}

// ---------------------------------------------------------------------------
// node update: self-interaction + aggregate linears + gated nonlinearity
// ---------------------------------------------------------------------------
template <bool SIV>
__global__ void k_node_update(float* __restrict__ s_cur, float* __restrict__ v_cur,
                              const float* __restrict__ agg_s, const float* __restrict__ agg_v,
                              const float* __restrict__ lin1_s, const float* __restrict__ lin1_v,
                              const float* __restrict__ lin2_s, const float* __restrict__ lin2_v) {
    int n = blockIdx.x * blockDim.x + threadIdx.x;
    if (n >= N_NODES) return;
    float sv[10], ag[20], su[20];
#pragma unroll
    for (int i = 0; i < 10; ++i) sv[i] = s_cur[n * 10 + i];
#pragma unroll
    for (int i = 0; i < 20; ++i) ag[i] = agg_s[n * 20 + i];
#pragma unroll
    for (int o = 0; o < 20; ++o) {
        float a = 0.f, b = 0.f;
#pragma unroll
        for (int i = 0; i < 10; ++i) a += sv[i] * lin1_s[i * 20 + o];
#pragma unroll
        for (int i = 0; i < 20; ++i) b += ag[i] * lin2_s[i * 20 + o];
        su[o] = SQ2 * (a * INV_SQRT10 + b * INV_SQRT20);
    }
    float av[30];
#pragma unroll
    for (int q = 0; q < 30; ++q) av[q] = agg_v[n * 30 + q];
    float vin[30];
    if (SIV) {
#pragma unroll
        for (int q = 0; q < 30; ++q) vin[q] = v_cur[n * 30 + q];
    }
#pragma unroll
    for (int o = 0; o < 10; ++o) {
        const float gate = sigmoidf(su[10 + o]);
#pragma unroll
        for (int m = 0; m < 3; ++m) {
            float xv = 0.f;
#pragma unroll
            for (int i = 0; i < 10; ++i) xv += av[i * 3 + m] * lin2_v[i * 10 + o];
            xv *= INV_SQRT10;
            float vvv;
            if (SIV) {
                float siv = 0.f;
#pragma unroll
                for (int i = 0; i < 10; ++i) siv += vin[i * 3 + m] * lin1_v[i * 10 + o];
                vvv = SQ2 * (siv * INV_SQRT10 + xv);
            } else {
                vvv = xv;   // mask==0 path: res_v = x_v
            }
            v_cur[n * 30 + o * 3 + m] = gate * vvv;
        }
    }
#pragma unroll
    for (int i = 0; i < 10; ++i) s_cur[n * 10 + i] = su[i] * sigmoidf(su[i]);
}

// final layer node output + graph readout (segment_sum via atomics)
__global__ void k_final(const float* __restrict__ s_cur, const float* __restrict__ agg1,
                        const int* __restrict__ batch, const float* __restrict__ lin1,
                        const float* __restrict__ lin2, float* __restrict__ out) {
    int n = blockIdx.x * blockDim.x + threadIdx.x;
    if (n >= N_NODES) return;
    float si = 0.f;
#pragma unroll
    for (int i = 0; i < 10; ++i) si += s_cur[n * 10 + i] * lin1[i];
    float hs = SQ2 * (si * INV_SQRT10 + agg1[n] * lin2[0]);
    atomicAdd(&out[batch[n]], hs);
}

// ---------------------------------------------------------------------------
template <int MS, int MV, int OUT_S, int OUT_V>
constexpr size_t fused_smem_bytes() {
    constexpr int MVp = MV > 0 ? MV : 1;
    constexpr int OVp = OUT_V > 0 ? OUT_V : 1;
    return (size_t)EPW * KP * 2 * 2                      // hA + hB (f16)
         + (size_t)EPW * 64 * 4                          // w chunk
         + (size_t)EPW * (MS + MVp + MS + MVp * 3 + OUT_S + OVp + OVp * 3 + 1) * 4;
}

extern "C" void kernel_launch(void* const* d_in, const int* in_sizes, int n_in,
                              void* d_out, int out_size, void* d_ws, size_t ws_size,
                              hipStream_t stream) {
    (void)in_sizes; (void)n_in; (void)out_size; (void)ws_size;
    const int*   z     = (const int*)d_in[0];
    const float* pos   = (const float*)d_in[1];
    const int*   eidx  = (const int*)d_in[2];
    const int*   batch = (const int*)d_in[3];
    const float* emb   = (const float*)d_in[4];
    const float *rW0[3], *rb0[3], *rW1[3], *rb1[3], *rW2[3], *rb2[3];
    for (int l = 0; l < 3; ++l) {
        rW0[l] = (const float*)d_in[5 + 6 * l + 0];
        rb0[l] = (const float*)d_in[5 + 6 * l + 1];
        rW1[l] = (const float*)d_in[5 + 6 * l + 2];
        rb1[l] = (const float*)d_in[5 + 6 * l + 3];
        rW2[l] = (const float*)d_in[5 + 6 * l + 4];
        rb2[l] = (const float*)d_in[5 + 6 * l + 5];
    }
    const float* l0_lin1_s = (const float*)d_in[23];
    const float* l0_lin2_s = (const float*)d_in[24];
    const float* l0_lin2_v = (const float*)d_in[25];
    const float* l1_lin1_s = (const float*)d_in[26];
    const float* l1_lin1_v = (const float*)d_in[27];
    const float* l1_lin2_s = (const float*)d_in[28];
    const float* l1_lin2_v = (const float*)d_in[29];
    const float* l2_lin1_s = (const float*)d_in[30];
    const float* l2_lin2_s = (const float*)d_in[31];

    // workspace layout
    char* p = (char*)d_ws;
    float* s_cur = (float*)p;  p += (size_t)N_NODES * 10 * 4;
    float* v_cur = (float*)p;  p += (size_t)N_NODES * 30 * 4;
    float* agg_s = (float*)p;  p += (size_t)N_NODES * 20 * 4;
    float* agg_v = (float*)p;  p += (size_t)N_NODES * 30 * 4;
    const int NWs[3]   = {300, 600, 20};
    const int NPAD2[3] = {320, 640, 64};
    _Float16* W1t[3];
    _Float16* W2t[3];
    for (int l = 0; l < 3; ++l) { W1t[l] = (_Float16*)p; p += (size_t)208 * KP * 2; }
    for (int l = 0; l < 3; ++l) { W2t[l] = (_Float16*)p; p += (size_t)NPAD2[l] * KP * 2; }

    // pack radial weights to f16 [N][Kpad]
    for (int l = 0; l < 3; ++l) {
        int tot1 = 208 * KP;
        k_pack_w<<<(tot1 + 255) / 256, 256, 0, stream>>>(rW1[l], W1t[l], RAD_H, RAD_H, 208);
        int tot2 = NPAD2[l] * KP;
        k_pack_w<<<(tot2 + 255) / 256, 256, 0, stream>>>(rW2[l], W2t[l], RAD_H, NWs[l], NPAD2[l]);
    }

    k_node_init<<<(N_NODES + 255) / 256, 256, 0, stream>>>(z, emb, s_cur);

    const int nAgg   = N_NODES * (20 + 30);   // agg_s + agg_v contiguous
    const int eGrid  = N_EDGES / EPW;         // 6250 exactly
    const int nGrid  = (N_NODES + 255) / 256;

    // ---- layer 0: s only input, out_s=20, out_v=10, NW=300 ----
    k_zero<<<(nAgg + 255) / 256, 256, 0, stream>>>(agg_s, nAgg);
    k_edge_fused<10, 0, 20, 10, 300, 5, 1, 1>
        <<<eGrid, 128, fused_smem_bytes<10, 0, 20, 10>(), stream>>>(
            pos, eidx, s_cur, v_cur, rW0[0], rb0[0], W1t[0], rb1[0], W2t[0], rb2[0],
            agg_s, agg_v);
    k_node_update<false><<<nGrid, 256, 0, stream>>>(s_cur, v_cur, agg_s, agg_v,
                                                    l0_lin1_s, nullptr, l0_lin2_s, l0_lin2_v);

    // ---- layer 1: s+v input, NW=600 ----
    k_zero<<<(nAgg + 255) / 256, 256, 0, stream>>>(agg_s, nAgg);
    k_edge_fused<10, 10, 20, 10, 600, 10, 2, 2>
        <<<eGrid, 128, fused_smem_bytes<10, 10, 20, 10>(), stream>>>(
            pos, eidx, s_cur, v_cur, rW0[1], rb0[1], W1t[1], rb1[1], W2t[1], rb2[1],
            agg_s, agg_v);
    k_node_update<true><<<nGrid, 256, 0, stream>>>(s_cur, v_cur, agg_s, agg_v,
                                                   l1_lin1_s, l1_lin1_v, l1_lin2_s, l1_lin2_v);

    // ---- layer 2: scalar output only, NW=20; then graph readout ----
    k_zero<<<(nAgg + 255) / 256, 256, 0, stream>>>(agg_s, nAgg);
    k_edge_fused<10, 10, 1, 0, 20, 1, 2, 1>
        <<<eGrid, 128, fused_smem_bytes<10, 10, 1, 0>(), stream>>>(
            pos, eidx, s_cur, v_cur, rW0[2], rb0[2], W1t[2], rb1[2], W2t[2], rb2[2],
            agg_s, agg_v);
    k_zero<<<1, 256, 0, stream>>>((float*)d_out, NGRAPH);
    k_final<<<nGrid, 256, 0, stream>>>(s_cur, agg_s, batch, l2_lin1_s, l2_lin2_s,
                                       (float*)d_out);
}
